// AudioAttNet_66236985639024
// MI455X (gfx1250) — matrix operand order repaired and checked by reading, not compiled, not run on hardware
//
#include <hip/hip_runtime.h>

// ---------------------------------------------------------------------------
// Fused AudioAttNet: conv1d x4 (+LeakyReLU) -> per-batch [8,128]x[128,128]^T
// matmul (WMMA f16->f32) -> softmax over seq (per-lane, using the 16x16 C
// layout) -> attention-weighted sum -> out[B,128] f32.
// One 256-thread block (8 wave32) processes 16 batch elements.
// ---------------------------------------------------------------------------

typedef __attribute__((ext_vector_type(16))) _Float16 v16h;
typedef __attribute__((ext_vector_type(8)))  float    v8f;
typedef __attribute__((ext_vector_type(4)))  float    v4f;

#define NEG_SLOPE 0.02f
#define BT        16         // batches per workgroup
#define NTHREADS  256

// LDS layout (element counts)
#define XL_N   (BT * 29 * 8)      // 3712 f32  : input tile
#define Y1_N   (BT * 16 * 8)      // 2048 f32  : conv1 out
#define Y2_N   (BT * 8 * 8)       // 1024 f32  : conv2 out
#define Y3_N   (BT * 4 * 8)       //  512 f32  : conv3 out
#define Y4_N   (BT * 8 * 128)     // 16384 f16 : conv4 out, [b][s][d] d-contig
#define WL_N   (128 * 128)        // 16384 f16 : linear weight, [e][d] d-contig
#define SMEM_BYTES (((XL_N + Y1_N + Y2_N + Y3_N) * 4) + ((Y4_N + WL_N) * 2))  // 94720 B

union V16 { v16h v; v4f q[2]; };

// acc[s] += k0*x[s-1] + k1*x[s] + k2*x[s+1]  ('SAME' zero padding, S=8)
__device__ __forceinline__ void conv3_acc(float acc[8], const float xv[8],
                                          float k0, float k1, float k2) {
#pragma unroll
  for (int s = 0; s < 8; ++s) {
    float v = xv[s] * k1;
    if (s > 0) v += xv[s - 1] * k0;
    if (s < 7) v += xv[s + 1] * k2;
    acc[s] += v;
  }
}

__device__ __forceinline__ float lrelu(float v) {
  return v >= 0.0f ? v : NEG_SLOPE * v;
}

__global__ __launch_bounds__(NTHREADS) void audio_attnet_fused(
    const float* __restrict__ x,
    const float* __restrict__ w1, const float* __restrict__ b1,
    const float* __restrict__ w2, const float* __restrict__ b2,
    const float* __restrict__ w3, const float* __restrict__ b3,
    const float* __restrict__ w4, const float* __restrict__ b4,
    const float* __restrict__ wl,
    float* __restrict__ out, int B) {
  extern __shared__ char smem_raw[];
  float*    xl  = (float*)smem_raw;
  float*    y1  = xl + XL_N;
  float*    y2  = y1 + Y1_N;
  float*    y3  = y2 + Y2_N;
  _Float16* y4  = (_Float16*)(y3 + Y3_N);
  _Float16* wlh = y4 + Y4_N;

  const int t      = threadIdx.x;
  const int wgBase = blockIdx.x * BT;

  // ---- Phase 0: cooperative loads (coalesced) --------------------------
  {
    const long xtot  = (long)B * 232;        // 29*8 floats per batch
    const long gbase = (long)wgBase * 232;
    for (int i = t; i < XL_N; i += NTHREADS) {
      long gi = gbase + i;
      xl[i] = (gi < xtot) ? x[gi] : 0.0f;
    }
    for (int i = t; i < WL_N; i += NTHREADS)
      wlh[i] = (_Float16)wl[i];              // f32 -> f16 for WMMA B operand
  }
  __syncthreads();

  // ---- Phase 1: conv1 29->16  (256 threads: one (b,oc) row each) -------
  {
    const int b  = t >> 4;       // 0..15
    const int oc = t & 15;       // 0..15
    float acc[8];
    const float bias = b1[oc];
#pragma unroll
    for (int s = 0; s < 8; ++s) acc[s] = bias;
    for (int ic = 0; ic < 29; ++ic) {
      const float* xr = xl + (b * 29 + ic) * 8;
      float xv[8];
#pragma unroll
      for (int s = 0; s < 8; ++s) xv[s] = xr[s];
      const float* wp = w1 + (oc * 29 + ic) * 3;
      conv3_acc(acc, xv, wp[0], wp[1], wp[2]);
    }
    float* yo = y1 + (b * 16 + oc) * 8;
#pragma unroll
    for (int s = 0; s < 8; ++s) yo[s] = lrelu(acc[s]);
  }
  __syncthreads();

  // ---- Phase 2: conv2 16->8  (128 active threads) ----------------------
  if (t < 128) {
    const int b  = t >> 3;
    const int oc = t & 7;
    float acc[8];
    const float bias = b2[oc];
#pragma unroll
    for (int s = 0; s < 8; ++s) acc[s] = bias;
    for (int ic = 0; ic < 16; ++ic) {
      const float* xr = y1 + (b * 16 + ic) * 8;
      float xv[8];
#pragma unroll
      for (int s = 0; s < 8; ++s) xv[s] = xr[s];
      const float* wp = w2 + (oc * 16 + ic) * 3;
      conv3_acc(acc, xv, wp[0], wp[1], wp[2]);
    }
    float* yo = y2 + (b * 8 + oc) * 8;
#pragma unroll
    for (int s = 0; s < 8; ++s) yo[s] = lrelu(acc[s]);
  }
  __syncthreads();

  // ---- Phase 3: conv3 8->4  (64 active threads) ------------------------
  if (t < 64) {
    const int b  = t >> 2;
    const int oc = t & 3;
    float acc[8];
    const float bias = b3[oc];
#pragma unroll
    for (int s = 0; s < 8; ++s) acc[s] = bias;
#pragma unroll
    for (int ic = 0; ic < 8; ++ic) {
      const float* xr = y2 + (b * 8 + ic) * 8;
      float xv[8];
#pragma unroll
      for (int s = 0; s < 8; ++s) xv[s] = xr[s];
      const float* wp = w3 + (oc * 8 + ic) * 3;
      conv3_acc(acc, xv, wp[0], wp[1], wp[2]);
    }
    float* yo = y3 + (b * 4 + oc) * 8;
#pragma unroll
    for (int s = 0; s < 8; ++s) yo[s] = lrelu(acc[s]);
  }
  __syncthreads();

  // ---- Phase 4: conv4 4->128, store f16 in [b][s][d] (A-fragment form) -
  {
    const int b   = t >> 4;            // 0..15
    const int oc0 = (t & 15) * 8;      // 8 output channels per thread
    float xin[4][8];
#pragma unroll
    for (int ic = 0; ic < 4; ++ic) {
      const float* xr = y3 + (b * 4 + ic) * 8;
#pragma unroll
      for (int s = 0; s < 8; ++s) xin[ic][s] = xr[s];
    }
    for (int ocl = 0; ocl < 8; ++ocl) {
      const int oc = oc0 + ocl;
      float acc[8];
      const float bias = b4[oc];
#pragma unroll
      for (int s = 0; s < 8; ++s) acc[s] = bias;
#pragma unroll
      for (int ic = 0; ic < 4; ++ic) {
        const float* wp = w4 + (oc * 4 + ic) * 3;
        conv3_acc(acc, xin[ic], wp[0], wp[1], wp[2]);
      }
#pragma unroll
      for (int s = 0; s < 8; ++s)
        y4[(b * 8 + s) * 128 + oc] = (_Float16)lrelu(acc[s]);
    }
  }
  __syncthreads();

  // ---- Phase 5: WMMA matmul + per-lane softmax + weighted sum ----------
  // Each wave handles 2 batches packed as the 16 M-rows of the A operand
  // (M 0..7 = batch0 seq 0..7, M 8..15 = batch1 seq 0..7).
  // Note: linear bias bl is constant over the softmax axis -> cancels.
  {
    const int wave     = t >> 5;
    const int lane     = t & 31;
    const int pairBase = wave * 2;

    // A fragments: Y[16 x 128] in 4 K-steps of 32 (f16 16x32 A layout:
    // lane<16 holds K {0..7,16..23}, lane>=16 holds K {8..15,24..31})
    const int m  = lane & 15;
    const int ab = pairBase + (m >> 3);
    const int sA = m & 7;
    const int kA = (lane < 16) ? 0 : 8;
    v16h afrag[4];
#pragma unroll
    for (int kk = 0; kk < 4; ++kk) {
      V16 u;
      const int base = (ab * 8 + sA) * 128 + kk * 32 + kA;
      u.q[0] = *(const v4f*)(y4 + base);        // K = k0+kA .. +7
      u.q[1] = *(const v4f*)(y4 + base + 16);   // K = k0+16+kA .. +7
      afrag[kk] = u.v;
    }

    const int n  = lane & 15;
    const int kB = (lane < 16) ? 0 : 8;         // f16 32x16 B layout
    const int ob = pairBase + (lane >> 4);      // batch this lane's C covers
    const int gb = wgBase + ob;

#pragma unroll
    for (int nt = 0; nt < 8; ++nt) {
      const int n0 = nt * 16;
      v8f c = {};
#pragma unroll
      for (int kk = 0; kk < 4; ++kk) {
        V16 u;
        const int base = (n0 + n) * 128 + kk * 32 + kB;
        u.q[0] = *(const v4f*)(wlh + base);     // K = k0+kB .. +7, k0+16+kB .. +7
        u.q[1] = *(const v4f*)(wlh + base + 16);
        c = __builtin_amdgcn_wmma_f32_16x16x32_f16(
            false, afrag[kk], false, u.v, (short)0, c, false, false);
      }
      // c[v] = logits[seq=v][n0+n] for batch ob: softmax over v in-lane.
      float mx = c[0];
#pragma unroll
      for (int i = 1; i < 8; ++i) mx = fmaxf(mx, c[i]);
      float e[8], sum = 0.0f;
#pragma unroll
      for (int i = 0; i < 8; ++i) { e[i] = __expf(c[i] - mx); sum += e[i]; }
      const float inv = 1.0f / sum;
      float acc = 0.0f;
#pragma unroll
      for (int i = 0; i < 8; ++i)
        acc += e[i] * (float)y4[(ob * 8 + i) * 128 + n0 + n];
      acc *= inv;
      if (gb < B) out[(long)gb * 128 + n0 + n] = acc;
    }
  }
}

extern "C" void kernel_launch(void* const* d_in, const int* in_sizes, int n_in,
                              void* d_out, int out_size, void* d_ws, size_t ws_size,
                              hipStream_t stream) {
  const float* x  = (const float*)d_in[0];
  const float* w1 = (const float*)d_in[1];
  const float* b1 = (const float*)d_in[2];
  const float* w2 = (const float*)d_in[3];
  const float* b2 = (const float*)d_in[4];
  const float* w3 = (const float*)d_in[5];
  const float* b3 = (const float*)d_in[6];
  const float* w4 = (const float*)d_in[7];
  const float* b4 = (const float*)d_in[8];
  const float* wl = (const float*)d_in[9];
  // d_in[10] = bl: constant along the softmax axis -> cancels, unused.
  float* out = (float*)d_out;

  const int B   = in_sizes[0] / (29 * 8);
  const int nwg = (B + BT - 1) / BT;

  (void)hipFuncSetAttribute((const void*)audio_attnet_fused,
                            hipFuncAttributeMaxDynamicSharedMemorySize,
                            SMEM_BYTES);
  audio_attnet_fused<<<nwg, NTHREADS, SMEM_BYTES, stream>>>(
      x, w1, b1, w2, b2, w3, b3, w4, b4, wl, out, B);
}
